// LightningIndexer_37271726194771
// MI455X (gfx1250) — matrix operand order repaired and checked by reading, not compiled, not run on hardware
//
#include <hip/hip_runtime.h>
#include <hip/hip_bf16.h>

// ---------------- problem constants (fixed by reference) ----------------
constexpr int Bb   = 2;
constexpr int SQ   = 4096;
constexpr int SKV  = 16384;
constexpr int Dm   = 2048;   // D_MODEL
constexpr int DC   = 512;    // D_C
constexpr int CI   = 64;     // C_I
constexpr int NH   = 4;      // N_H_I
constexpr int RATE = 4;      // compressed_rate
constexpr int SK   = SKV / RATE;     // 4096
constexpr int MQ   = Bb * SQ;        // 8192 query rows
constexpr int MK   = Bb * SK;        // 8192 compressed kv rows
constexpr int TOPK = 512;
constexpr int SEG  = Bb * SQ * TOPK; // elements per output segment

// ---------------- workspace byte offsets ----------------
constexpr size_t WS_HQBF   = 0;                                   // MQ*Dm u16
constexpr size_t WS_KPOOL  = WS_HQBF  + (size_t)MQ * Dm * 2;      // MK*Dm u16
constexpr size_t WS_CQBF   = WS_KPOOL + (size_t)MK * Dm * 2;      // MQ*DC u16
constexpr size_t WS_QIBF   = WS_CQBF  + (size_t)MQ * DC * 2;      // MQ*NH*CI u16
constexpr size_t WS_KPROJ  = WS_QIBF  + (size_t)MQ * NH * CI * 2; // MK*CI u16
constexpr size_t WS_WF32   = WS_KPROJ + (size_t)MK * CI * 2;      // MQ*NH f32
constexpr size_t WS_WDQT   = WS_WF32  + (size_t)MQ * NH * 4;      // DC*Dm u16
constexpr size_t WS_WIUQT  = WS_WDQT  + (size_t)DC * Dm * 2;      // (NH*CI)*DC u16
constexpr size_t WS_WKT    = WS_WIUQT + (size_t)NH * CI * DC * 2; // CI*Dm u16

// ---------------- types ----------------
typedef __attribute__((ext_vector_type(16))) __bf16 v16bf;
typedef __attribute__((ext_vector_type(8)))  float  v8f;

union Frag16 { unsigned short us[16]; v16bf v; };

__device__ __forceinline__ unsigned short f2bf(float x) {
  unsigned u = __float_as_uint(x);
  unsigned r = u + 0x7FFFu + ((u >> 16) & 1u);      // round-to-nearest-even
  if ((u & 0x7F800000u) == 0x7F800000u) r = u;      // inf/nan passthrough
  return (unsigned short)(r >> 16);
}

// A fragment: 16x32 bf16, M = lane&15, K-pairs per ISA 7.12.2 layout.
__device__ __forceinline__ v16bf load_a_frag(const unsigned short* A, int lda,
                                             int m0, int kb, int lane) {
  const int m = lane & 15, half = lane >> 4;
  const unsigned short* row = A + (size_t)(m0 + m) * lda + kb;
  Frag16 f;
#pragma unroll
  for (int v = 0; v < 8; ++v) {
    const int k0 = ((v >> 2) << 4) + (half << 3) + ((v & 3) << 1);
    f.us[2 * v]     = row[k0];
    f.us[2 * v + 1] = row[k0 + 1];
  }
  return f.v;
}

// B fragment from transposed storage BT[n][k]: N = lane&15, K-pairs per ISA 7.12.2 layout (same as A).
__device__ __forceinline__ v16bf load_bT_frag(const unsigned short* BT, int ldb,
                                              int n0, int kb, int lane) {
  const int n = lane & 15, half = lane >> 4;
  const unsigned short* row = BT + (size_t)(n0 + n) * ldb + kb;
  Frag16 f;
#pragma unroll
  for (int v = 0; v < 8; ++v) {
    const int k0 = ((v >> 2) << 4) + (half << 3) + ((v & 3) << 1);
    f.us[2 * v]     = row[k0];
    f.us[2 * v + 1] = row[k0 + 1];
  }
  return f.v;
}

// ---------------- conversion / pooling kernels ----------------
__global__ void convert_weights_kernel(const float* Wdq, const float* Wiuq, const float* Wk,
                                       unsigned short* WdqT, unsigned short* WiuqT,
                                       unsigned short* WkT) {
  const int N1 = Dm * DC, N2 = DC * (NH * CI), N3 = Dm * CI;
  int i = blockIdx.x * blockDim.x + threadIdx.x;
  if (i < N1) {
    int k = i / DC, n = i % DC;
    WdqT[(size_t)n * Dm + k] = f2bf(Wdq[i]);
  } else if (i < N1 + N2) {
    int t = i - N1;
    int k = t / (NH * CI), n = t % (NH * CI);
    WiuqT[(size_t)n * DC + k] = f2bf(Wiuq[t]);
  } else if (i < N1 + N2 + N3) {
    int t = i - N1 - N2;
    int k = t / CI, n = t % CI;
    WkT[(size_t)n * Dm + k] = f2bf(Wk[t]);
  }
}

__global__ void convert_hq_kernel(const float* hq, unsigned short* hq_bf) {
  size_t i = (size_t)blockIdx.x * blockDim.x + threadIdx.x;
  if (i < (size_t)MQ * Dm) hq_bf[i] = f2bf(hq[i]);
}

// mean-pool h_kv over RATE-row windows -> bf16
__global__ void pool_kernel(const float* hkv, unsigned short* kpool) {
  size_t i = (size_t)blockIdx.x * blockDim.x + threadIdx.x;
  if (i >= (size_t)MK * Dm) return;
  size_t r = i / Dm, d = i % Dm;
  float s = 0.f;
#pragma unroll
  for (int j = 0; j < RATE; ++j) s += hkv[(r * RATE + j) * Dm + d];
  kpool[i] = f2bf(s * (1.0f / RATE));
}

// w = h_q @ W_w  (tiny GEMV, one thread per output)
__global__ void w_kernel(const float* hq, const float* Ww, float* wout) {
  int i = blockIdx.x * blockDim.x + threadIdx.x;
  if (i >= MQ * NH) return;
  int r = i / NH, h = i % NH;
  const float* row = hq + (size_t)r * Dm;
  float s = 0.f;
  for (int d = 0; d < Dm; ++d) s += row[d] * Ww[d * NH + h];
  wout[i] = s;
}

// ---------------- WMMA bf16 GEMM: 1 wave computes a 32x64 tile ----------------
// Double-buffered K loop: preload next 6 fragments, then issue 8 WMMAs.
__global__ __launch_bounds__(32)
void wmma_gemm_bf16(const unsigned short* __restrict__ A,
                    const unsigned short* __restrict__ BT,
                    float* Cf, unsigned short* Cbf,
                    int K, int lda, int ldb, int ldc) {
  const int lane = threadIdx.x & 31;
  const int m0 = blockIdx.x * 32;
  const int n0 = blockIdx.y * 64;
  v8f acc[2][4] = {};

  v16bf a0 = load_a_frag(A, lda, m0,      0, lane);
  v16bf a1 = load_a_frag(A, lda, m0 + 16, 0, lane);
  v16bf b[4];
#pragma unroll
  for (int t = 0; t < 4; ++t) b[t] = load_bT_frag(BT, ldb, n0 + 16 * t, 0, lane);

  for (int kb = 32; kb < K; kb += 32) {
    v16bf na0 = load_a_frag(A, lda, m0,      kb, lane);
    v16bf na1 = load_a_frag(A, lda, m0 + 16, kb, lane);
    v16bf nb[4];
#pragma unroll
    for (int t = 0; t < 4; ++t) nb[t] = load_bT_frag(BT, ldb, n0 + 16 * t, kb, lane);
#pragma unroll
    for (int t = 0; t < 4; ++t) {
      acc[0][t] = __builtin_amdgcn_wmma_f32_16x16x32_bf16(
          false, a0, false, b[t], (short)0, acc[0][t], false, false);
      acc[1][t] = __builtin_amdgcn_wmma_f32_16x16x32_bf16(
          false, a1, false, b[t], (short)0, acc[1][t], false, false);
    }
    a0 = na0; a1 = na1;
#pragma unroll
    for (int t = 0; t < 4; ++t) b[t] = nb[t];
  }
#pragma unroll
  for (int t = 0; t < 4; ++t) {
    acc[0][t] = __builtin_amdgcn_wmma_f32_16x16x32_bf16(
        false, a0, false, b[t], (short)0, acc[0][t], false, false);
    acc[1][t] = __builtin_amdgcn_wmma_f32_16x16x32_bf16(
        false, a1, false, b[t], (short)0, acc[1][t], false, false);
  }

  // C/D layout: N = lane&15, M = (lane>>4)*8 + j
  const int n = lane & 15, mb = (lane >> 4) * 8;
#pragma unroll
  for (int half = 0; half < 2; ++half)
#pragma unroll
    for (int t = 0; t < 4; ++t)
#pragma unroll
      for (int j = 0; j < 8; ++j) {
        const size_t o =
            (size_t)(m0 + 16 * half + mb + j) * ldc + (n0 + 16 * t + n);
        const float val = acc[half][t][j];
        if (Cf)  Cf[o]  = val;
        if (Cbf) Cbf[o] = f2bf(val);
      }
}

// ---------------- fused scores + top-k ----------------
// One block handles 16 query rows. A fragments (q_i), head weights and mask
// thresholds are loop-invariant per wave -> hoisted into registers. Inner loop:
// 2 B-fragment loads (k_proj, streamed once) + 8 WMMAs per 16x16 k-tile.
// Then exact sorted top-512 per row via bitonic sort in LDS.
constexpr size_t SMEM_BYTES = (size_t)16 * SK * sizeof(float) + (size_t)SK * 8;

__global__ __launch_bounds__(256)
void scores_topk_kernel(const unsigned short* __restrict__ qi,
                        const unsigned short* __restrict__ kproj,
                        const float* __restrict__ wgt,
                        const int* __restrict__ qpos,
                        float* __restrict__ out_scores,
                        int* __restrict__ out_idx) {
  extern __shared__ char smem[];
  float* s_scores = (float*)smem;                                    // 16*SK f32
  unsigned long long* s_sort =
      (unsigned long long*)(smem + (size_t)16 * SK * sizeof(float)); // SK u64

  const int tile = blockIdx.x;            // 0..MQ/16-1
  const int b    = (tile * 16) / SQ;
  const int q0   = (tile * 16) % SQ;
  const int bq0  = b * SQ + q0;
  const int lane = threadIdx.x & 31;
  const int wave = threadIdx.x >> 5;
  const int n = lane & 15, mb = (lane >> 4) * 8;
  const int div = SK / SQ + 1;
  const int SKT = SK / 16;                // 256 k-tiles

  // ---- hoist loop-invariant per-lane state ----
  v16bf afr[NH][2];
#pragma unroll
  for (int h = 0; h < NH; ++h)
#pragma unroll
    for (int c = 0; c < 2; ++c)
      afr[h][c] = load_a_frag(qi, NH * CI, bq0, h * CI + 32 * c, lane);

  float wv[8][NH];
  int   qd[8];
#pragma unroll
  for (int j = 0; j < 8; ++j) {
    const int m = mb + j;
    const float* wr = wgt + (size_t)(bq0 + m) * NH;
#pragma unroll
    for (int h = 0; h < NH; ++h) wv[j][h] = wr[h];
    qd[j] = qpos[bq0 + m] / div;
  }

  // ---- phase 1: k-tile loop, double-buffered B fragments ----
  const unsigned short* kp = kproj + (size_t)b * SK * CI;
  int kt = wave;
  v16bf b0 = load_bT_frag(kp, CI, kt * 16, 0, lane);
  v16bf b1 = load_bT_frag(kp, CI, kt * 16, 32, lane);
  while (true) {
    const int ktn = kt + 8;
    v16bf nb0 = b0, nb1 = b1;
    if (ktn < SKT) {
      nb0 = load_bT_frag(kp, CI, ktn * 16, 0, lane);
      nb1 = load_bT_frag(kp, CI, ktn * 16, 32, lane);
      if (ktn + 8 < SKT)  // NT-ish prefetch of the streamed k_proj panel
        __builtin_prefetch(kp + (size_t)(ktn + 8) * 16 * CI, 0, 1);
    }
    v8f acc[NH] = {};
#pragma unroll
    for (int h = 0; h < NH; ++h) {
      acc[h] = __builtin_amdgcn_wmma_f32_16x16x32_bf16(
          false, afr[h][0], false, b0, (short)0, acc[h], false, false);
      acc[h] = __builtin_amdgcn_wmma_f32_16x16x32_bf16(
          false, afr[h][1], false, b1, (short)0, acc[h], false, false);
    }
    const int kg = kt * 16 + n;
#pragma unroll
    for (int j = 0; j < 8; ++j) {
      float s = 0.f;
#pragma unroll
      for (int h = 0; h < NH; ++h) {
        const float d = acc[h][j];
        s += wv[j][h] * (d > 0.f ? d : 0.f);
      }
      s_scores[(mb + j) * SK + kg] = (kg < qd[j]) ? s : -__builtin_inff();
    }
    if (ktn >= SKT) break;
    b0 = nb0; b1 = nb1; kt = ktn;
  }
  __syncthreads();

  // ---- phase 2: exact sorted top-512 per row (bitonic sort of packed u64) ----
  for (int m = 0; m < 16; ++m) {
    for (int i = threadIdx.x; i < SK; i += 256) {
      const unsigned u = __float_as_uint(s_scores[m * SK + i]);
      const unsigned key = (u & 0x80000000u) ? ~u : (u ^ 0x80000000u);
      // descending by value; ties -> smaller index first
      s_sort[i] = ((unsigned long long)key << 32) |
                  (unsigned long long)(0xFFFFFFFFu - (unsigned)i);
    }
    __syncthreads();
    for (int size = 2; size <= SK; size <<= 1) {
      for (int stride = size >> 1; stride > 0; stride >>= 1) {
        for (int i = threadIdx.x; i < SK; i += 256) {
          const int j = i ^ stride;
          if (j > i) {
            const unsigned long long x = s_sort[i], y = s_sort[j];
            const bool desc = ((i & size) == 0);
            if (desc ? (x < y) : (x > y)) { s_sort[i] = y; s_sort[j] = x; }
          }
        }
        __syncthreads();
      }
    }
    for (int i = threadIdx.x; i < TOPK; i += 256) {
      const unsigned long long e = s_sort[i];
      const unsigned key = (unsigned)(e >> 32);
      const unsigned idx = 0xFFFFFFFFu - (unsigned)(e & 0xFFFFFFFFull);
      const unsigned ub = (key & 0x80000000u) ? (key ^ 0x80000000u) : ~key;
      const size_t o = (size_t)(bq0 + m) * TOPK + i;
      out_scores[o] = __uint_as_float(ub);
      out_idx[o]    = (int)idx;
    }
    __syncthreads();
  }
}

// ---------------- host launcher ----------------
extern "C" void kernel_launch(void* const* d_in, const int* in_sizes, int n_in,
                              void* d_out, int out_size, void* d_ws, size_t ws_size,
                              hipStream_t stream) {
  const float* h_q   = (const float*)d_in[0];
  const float* h_kv  = (const float*)d_in[1];
  const int*   qpos  = (const int*)d_in[2];
  // d_in[3] = compressed_rate (fixed at 4 per reference constants)
  const float* W_dq  = (const float*)d_in[4];
  const float* W_iuq = (const float*)d_in[5];
  const float* W_w   = (const float*)d_in[6];
  const float* W_k   = (const float*)d_in[7];
  (void)in_sizes; (void)n_in; (void)out_size; (void)ws_size;

  char* ws = (char*)d_ws;
  unsigned short* hq_bf   = (unsigned short*)(ws + WS_HQBF);
  unsigned short* kpool   = (unsigned short*)(ws + WS_KPOOL);
  unsigned short* cq_bf   = (unsigned short*)(ws + WS_CQBF);
  unsigned short* qi_bf   = (unsigned short*)(ws + WS_QIBF);
  unsigned short* kproj   = (unsigned short*)(ws + WS_KPROJ);
  float*          w_f32   = (float*)(ws + WS_WF32);
  unsigned short* WdqT    = (unsigned short*)(ws + WS_WDQT);
  unsigned short* WiuqT   = (unsigned short*)(ws + WS_WIUQT);
  unsigned short* WkT     = (unsigned short*)(ws + WS_WKT);

  float* out_scores = (float*)d_out;
  int*   out_idx    = ((int*)d_out) + SEG;
  float* out_cq     = ((float*)d_out) + 2 * (size_t)SEG;

  // 1) weight transpose+convert
  {
    const int tot = Dm * DC + DC * (NH * CI) + Dm * CI;
    convert_weights_kernel<<<(tot + 255) / 256, 256, 0, stream>>>(
        W_dq, W_iuq, W_k, WdqT, WiuqT, WkT);
  }
  // 2) h_q -> bf16
  convert_hq_kernel<<<(int)(((size_t)MQ * Dm + 255) / 256), 256, 0, stream>>>(h_q, hq_bf);
  // 3) head weights w
  w_kernel<<<(MQ * NH + 255) / 256, 256, 0, stream>>>(h_q, W_w, w_f32);
  // 4) mean-pool kv -> bf16
  pool_kernel<<<(int)(((size_t)MK * Dm + 255) / 256), 256, 0, stream>>>(h_kv, kpool);
  // 5) k_proj = pooled_kv @ W_k   (M=8192, N=64, K=2048)
  wmma_gemm_bf16<<<dim3(MK / 32, CI / 64), 32, 0, stream>>>(
      kpool, WkT, nullptr, kproj, Dm, Dm, Dm, CI);
  // 6) c_q = h_q @ W_dq           (M=8192, N=512, K=2048) -> fp32 out + bf16 ws
  wmma_gemm_bf16<<<dim3(MQ / 32, DC / 64), 32, 0, stream>>>(
      hq_bf, WdqT, out_cq, cq_bf, Dm, Dm, Dm, DC);
  // 7) q_i = c_q @ W_iuq          (M=8192, N=256, K=512)
  wmma_gemm_bf16<<<dim3(MQ / 32, (NH * CI) / 64), 32, 0, stream>>>(
      cq_bf, WiuqT, nullptr, qi_bf, DC, DC, DC, NH * CI);
  // 8) fused scores + exact sorted top-k (256 KB score tile + 32 KB sort in LDS)
  (void)hipFuncSetAttribute((const void*)scores_topk_kernel,
                            hipFuncAttributeMaxDynamicSharedMemorySize,
                            (int)SMEM_BYTES);
  scores_topk_kernel<<<MQ / 16, 256, SMEM_BYTES, stream>>>(
      qi_bf, kproj, w_f32, qpos, out_scores, out_idx);
}
